// GATModel_70970039599394
// MI455X (gfx1250) — matrix operand order repaired and checked by reading, not compiled, not run on hardware
//
#include <hip/hip_runtime.h>
#include <hip/hip_bf16.h>

#define N_NODES 50000
#define N_EDGES 800000
#define F_IN    128
#define NHEAD   4
#define DHEAD   64
#define HDIM    256          // NHEAD * DHEAD
#define NCLS    40
#define NCLS_PAD 48
#define EP      (N_EDGES + N_NODES)   // edges incl. self loops
#define NTM     (N_NODES / 16)        // 3125 row tiles

typedef __attribute__((ext_vector_type(16))) _Float16 v16h;
typedef __attribute__((ext_vector_type(8)))  float    v8f;

// ---------------------------------------------------------------- packing
// Fuse f32->f16 conversion with reordering into the wave32 WMMA fragment
// layout (ISA 7.12.2), so the GEMM reads contiguous 32B per lane.
//
// A fragment, 16x32 f16: lane L holds row M=L%16; half element e (0..15):
//   v = e/2, k = (v/4)*16 + (L/16)*8 + (v%2==..)*..  (see formula below)
// Packed A element index: ((tm*KS + ks)*32 + lane)*16 + e
__global__ void pack_a_kernel(const float* __restrict__ X,
                              _Float16* __restrict__ Ap, int K) {
    int KS = K >> 5;
    int idx = blockIdx.x * blockDim.x + threadIdx.x;
    int total = NTM * KS * 512;                  // 32 lanes * 16 halves
    if (idx >= total) return;
    int e    = idx & 15;
    int lane = (idx >> 4) & 31;
    int ts   = idx >> 9;                         // tm*KS + ks
    int ks   = ts % KS;
    int tm   = ts / KS;
    int v  = e >> 1;
    int kk = (v >> 2) * 16 + (lane >> 4) * 8 + (v & 3) * 2 + (e & 1);
    int m  = tm * 16 + (lane & 15);
    int k  = ks * 32 + kk;
    Ap[idx] = (_Float16)X[(size_t)m * K + k];
}

// B fragment, 32x16 f16: lane L holds column N=L%16; half h -> K=(h/8)*16+(L/16)*8+h%8.
// Packed B element index: ((tn*KS + ks)*32 + lane)*16 + h.
// NcSrc < Nc handles zero-padding of the classifier weights (40 -> 48).
__global__ void pack_b_kernel(const float* __restrict__ W,
                              _Float16* __restrict__ Bp,
                              int K, int Nc, int NcSrc) {
    int KS = K >> 5;
    int idx = blockIdx.x * blockDim.x + threadIdx.x;
    int total = (Nc / 16) * KS * 512;
    if (idx >= total) return;
    int h    = idx & 15;
    int lane = (idx >> 4) & 31;
    int ts   = idx >> 9;
    int ks   = ts % KS;
    int tn   = ts / KS;
    int k = ks * 32 + (h >> 3) * 16 + (lane >> 4) * 8 + (h & 7);
    int n = tn * 16 + (lane & 15);
    Bp[idx] = (n < NcSrc) ? (_Float16)W[(size_t)k * NcSrc + n] : (_Float16)0.0f;
}

// ---------------------------------------------------------------- WMMA GEMM
// C[M,Nc] = A * B from packed fragments. Each wave computes a 16 x (16*NT)
// strip: one A fragment feeds NT WMMAs per 32-wide K step.
template <int NT>
__global__ void wmma_gemm_packed_kernel(const v16h* __restrict__ Ap,
                                        const v16h* __restrict__ Bp,
                                        float* __restrict__ C, int K, int Nc) {
    int KS   = K >> 5;
    int ngrp = Nc / (16 * NT);
    int wid  = threadIdx.x >> 5;
    int lane = threadIdx.x & 31;
    int wg   = blockIdx.x * 4 + wid;
    if (wg >= NTM * ngrp) return;                // wave-uniform exit
    int tm = wg / ngrp, tg = wg % ngrp;

    v8f acc[NT];
#pragma unroll
    for (int t = 0; t < NT; ++t) acc[t] = (v8f){};

    const v16h* ap = Ap + (size_t)tm * KS * 32 + lane;
    const v16h* bp = Bp + (size_t)(tg * NT) * KS * 32 + lane;
    for (int ks = 0; ks < KS; ++ks) {
        v16h a = ap[(size_t)ks * 32];
#pragma unroll
        for (int t = 0; t < NT; ++t) {
            v16h b = bp[((size_t)t * KS + ks) * 32];
            acc[t] = __builtin_amdgcn_wmma_f32_16x16x32_f16(
                false, a, false, b, (short)0, acc[t], false, false);
        }
    }
    int m_in = lane & 15, hi = lane >> 4;
#pragma unroll
    for (int t = 0; t < NT; ++t) {
        int n0 = (tg * NT + t) * 16 + m_in;
#pragma unroll
        for (int r = 0; r < 8; ++r)
            C[(size_t)(tm * 16 + r + hi * 8) * Nc + n0] = acc[t][r];
    }
}

// ---------------------------------------------------------------- utilities
__global__ void fill_f32_kernel(float* __restrict__ p, float v, int n) {
    int i = blockIdx.x * blockDim.x + threadIdx.x;
    if (i < n) p[i] = v;
}

// float atomic max via int/uint monotonic bit trick (values never NaN here)
__device__ inline void atomicMaxFloat(float* addr, float val) {
    if (!(val < 0.0f)) atomicMax((int*)addr, __float_as_int(val));
    else               atomicMin((unsigned int*)addr, __float_as_uint(val));
}

// ---------------------------------------------------------------- GAT pieces
__global__ void alpha_kernel(const float* __restrict__ h,
                             const float* __restrict__ a_src,
                             const float* __restrict__ a_dst,
                             float* __restrict__ as_out,
                             float* __restrict__ ad_out) {
    int idx = blockIdx.x * blockDim.x + threadIdx.x;   // n*NHEAD + head
    if (idx >= N_NODES * NHEAD) return;
    int n = idx / NHEAD, hh = idx % NHEAD;
    const float* hp = h + (size_t)n * HDIM + hh * DHEAD;
    const float* ps = a_src + hh * DHEAD;
    const float* pd = a_dst + hh * DHEAD;
    float accs = 0.f, accd = 0.f;
#pragma unroll 8
    for (int d = 0; d < DHEAD; ++d) {
        float v = hp[d];
        accs += v * ps[d];
        accd += v * pd[d];
    }
    as_out[idx] = accs;
    ad_out[idx] = accd;
}

__device__ inline void edge_endpoints(const long long* ei, long long eid,
                                      int& src, int& dst) {
    if (eid < N_EDGES) {
        src = (int)ei[eid];
        dst = (int)ei[N_EDGES + eid];
    } else {
        src = dst = (int)(eid - N_EDGES);     // self loop
    }
}

__global__ void edge_max_kernel(const long long* __restrict__ ei,
                                const float* __restrict__ as,
                                const float* __restrict__ ad,
                                float* __restrict__ ebuf,
                                float* __restrict__ m) {
    int idx = blockIdx.x * blockDim.x + threadIdx.x;   // over EP*NHEAD
    if (idx >= EP * NHEAD) return;
    long long eid = idx / NHEAD;
    int hh = idx % NHEAD;
    int src, dst;
    edge_endpoints(ei, eid, src, dst);
    float ev = as[src * NHEAD + hh] + ad[dst * NHEAD + hh];
    ev = ev > 0.f ? ev : 0.2f * ev;
    ebuf[idx] = ev;
    atomicMaxFloat(&m[dst * NHEAD + hh], ev);
}

__global__ void edge_expsum_kernel(const long long* __restrict__ ei,
                                   float* __restrict__ ebuf,
                                   const float* __restrict__ m,
                                   float* __restrict__ ssum) {
    int idx = blockIdx.x * blockDim.x + threadIdx.x;
    if (idx >= EP * NHEAD) return;
    long long eid = idx / NHEAD;
    int hh = idx % NHEAD;
    int src, dst;
    edge_endpoints(ei, eid, src, dst);
    float ex = expf(ebuf[idx] - m[dst * NHEAD + hh]);
    ebuf[idx] = ex;
    atomicAdd(&ssum[dst * NHEAD + hh], ex);
}

// one block per edge, one lane per feature: coalesced 1KB gather + scatter
__global__ void edge_aggregate_kernel(const long long* __restrict__ ei,
                                      const float* __restrict__ ebuf,
                                      const float* __restrict__ ssum,
                                      const float* __restrict__ h,
                                      float* __restrict__ out) {
    long long eid = blockIdx.x;
    int t = threadIdx.x;                    // 0..255 => (head, d)
    int hh = t >> 6;                        // DHEAD = 64
    int src, dst;
    edge_endpoints(ei, eid, src, dst);
    float ex  = ebuf[eid * NHEAD + hh];
    float att = ex / (ssum[dst * NHEAD + hh] + 1e-16f);
    float val = h[(size_t)src * HDIM + t] * att;
    atomicAdd(&out[(size_t)dst * HDIM + t], val);
}

__global__ void bias_elu_kernel(float* __restrict__ out,
                                const float* __restrict__ b, int n) {
    int i = blockIdx.x * blockDim.x + threadIdx.x;
    if (i >= n) return;
    float v = out[i] + b[i % HDIM];
    out[i] = v > 0.f ? v : (expf(v) - 1.0f);
}

__global__ void logsoftmax_kernel(const float* __restrict__ logits,
                                  const float* __restrict__ bl,
                                  float* __restrict__ out) {
    int n = blockIdx.x * blockDim.x + threadIdx.x;
    if (n >= N_NODES) return;
    float v[NCLS];
    float mx = -INFINITY;
#pragma unroll
    for (int c = 0; c < NCLS; ++c) {
        v[c] = logits[(size_t)n * NCLS_PAD + c] + bl[c];
        mx = fmaxf(mx, v[c]);
    }
    float s = 0.f;
#pragma unroll
    for (int c = 0; c < NCLS; ++c) s += expf(v[c] - mx);
    float lse = mx + logf(s);
#pragma unroll
    for (int c = 0; c < NCLS; ++c) out[(size_t)n * NCLS + c] = v[c] - lse;
}

// ---------------------------------------------------------------- launcher
extern "C" void kernel_launch(void* const* d_in, const int* in_sizes, int n_in,
                              void* d_out, int out_size, void* d_ws, size_t ws_size,
                              hipStream_t stream) {
    const float*     x  = (const float*)d_in[0];
    const long long* ei = (const long long*)d_in[1];
    const float* W[3]  = {(const float*)d_in[2], (const float*)d_in[6],  (const float*)d_in[10]};
    const float* As[3] = {(const float*)d_in[3], (const float*)d_in[7],  (const float*)d_in[11]};
    const float* Ad[3] = {(const float*)d_in[4], (const float*)d_in[8],  (const float*)d_in[12]};
    const float* Bv[3] = {(const float*)d_in[5], (const float*)d_in[9],  (const float*)d_in[13]};
    const float* Wl = (const float*)d_in[14];
    const float* bl = (const float*)d_in[15];

    // workspace carve
    char* ws = (char*)d_ws;
    size_t off = 0;
    auto carve = [&](size_t bytes) -> void* {
        void* p = ws + off;
        off = (off + bytes + 255) & ~(size_t)255;
        return p;
    };
    _Float16* Ap   = (_Float16*)carve((size_t)N_NODES * HDIM * sizeof(_Float16));
    _Float16* Bp   = (_Float16*)carve((size_t)HDIM * HDIM * sizeof(_Float16));
    float* bufG    = (float*)carve((size_t)N_NODES * HDIM * sizeof(float)); // h after GEMM
    float* bufX    = (float*)carve((size_t)N_NODES * HDIM * sizeof(float)); // layer in/out
    float* alpS    = (float*)carve((size_t)N_NODES * NHEAD * sizeof(float));
    float* alpD    = (float*)carve((size_t)N_NODES * NHEAD * sizeof(float));
    float* mbuf    = (float*)carve((size_t)N_NODES * NHEAD * sizeof(float));
    float* sbuf    = (float*)carve((size_t)N_NODES * NHEAD * sizeof(float));
    float* ebuf    = (float*)carve((size_t)EP * NHEAD * sizeof(float));
    float* lg      = (float*)carve((size_t)N_NODES * NCLS_PAD * sizeof(float));

    const int T = 256;
    const int nhN = N_NODES * NHEAD;
    const int ehN = EP * NHEAD;
    const int hdN = N_NODES * HDIM;

    for (int l = 0; l < 3; ++l) {
        int K = (l == 0) ? F_IN : HDIM;
        const float* xin = (l == 0) ? x : bufX;
        int KS = K / 32;

        // 1) pack activations + weights into WMMA fragment layout (f16)
        int aTot = NTM * KS * 512;
        pack_a_kernel<<<(aTot + T - 1) / T, T, 0, stream>>>(xin, Ap, K);
        int bTot = (HDIM / 16) * KS * 512;
        pack_b_kernel<<<(bTot + T - 1) / T, T, 0, stream>>>(W[l], Bp, K, HDIM, HDIM);

        // 2) h = x @ W  (WMMA, 16x64 strip per wave)
        {
            int waves = NTM * (HDIM / 64);     // ngrp = Nc/(16*4)
            wmma_gemm_packed_kernel<4><<<(waves + 3) / 4, 128, 0, stream>>>(
                (const v16h*)Ap, (const v16h*)Bp, bufG, K, HDIM);
        }

        // 3) attention coefficients per node/head
        alpha_kernel<<<(nhN + T - 1) / T, T, 0, stream>>>(bufG, As[l], Ad[l], alpS, alpD);

        // 4) init reductions and output accumulator (xin already consumed)
        fill_f32_kernel<<<(nhN + T - 1) / T, T, 0, stream>>>(mbuf, -INFINITY, nhN);
        fill_f32_kernel<<<(nhN + T - 1) / T, T, 0, stream>>>(sbuf, 0.0f, nhN);
        fill_f32_kernel<<<(hdN + T - 1) / T, T, 0, stream>>>(bufX, 0.0f, hdN);

        // 5) segment softmax over destination nodes
        edge_max_kernel<<<(ehN + T - 1) / T, T, 0, stream>>>(ei, alpS, alpD, ebuf, mbuf);
        edge_expsum_kernel<<<(ehN + T - 1) / T, T, 0, stream>>>(ei, ebuf, mbuf, sbuf);

        // 6) weighted scatter-aggregate
        edge_aggregate_kernel<<<EP, 256, 0, stream>>>(ei, ebuf, sbuf, bufG, bufX);

        // 7) bias + ELU
        bias_elu_kernel<<<(hdN + T - 1) / T, T, 0, stream>>>(bufX, Bv[l], hdN);
    }

    // final linear head (padded to 48 cols) + log_softmax
    {
        int KS = HDIM / 32;
        int aTot = NTM * KS * 512;
        pack_a_kernel<<<(aTot + T - 1) / T, T, 0, stream>>>(bufX, Ap, HDIM);
        int bTot = (NCLS_PAD / 16) * KS * 512;
        pack_b_kernel<<<(bTot + T - 1) / T, T, 0, stream>>>(Wl, Bp, HDIM, NCLS_PAD, NCLS);
        int waves = NTM;                        // ngrp = 48/(16*3) = 1
        wmma_gemm_packed_kernel<3><<<(waves + 3) / 4, 128, 0, stream>>>(
            (const v16h*)Ap, (const v16h*)Bp, lg, HDIM, NCLS_PAD);
    }
    logsoftmax_kernel<<<(N_NODES + T - 1) / T, T, 0, stream>>>(lg, bl, (float*)d_out);
}
